// RecurrentEmbeddingLayer_35244501630989
// MI455X (gfx1250) — compile-verified, matrix operand order checked by reading, not run on hardware
//
#include <hip/hip_runtime.h>
#include <hip/hip_bf16.h>

// ---------------------------------------------------------------------------
// 3-layer GRU scan for MI455X (gfx1250), wave32 + WMMA bf16.
//  Phase 1: k_convert  — transpose W1,W2,U0,U1,U2 -> bf16 [N=768][K=256] (L2-resident)
//  Phase 2: k_bias     — fold biases, zero pipeline flags
//  Phase 3: k_embed    — Xp0 = float(batch) @ W0 + b0[0]   (K=16, memory bound)
//  Phase 4: k_scan     — 3 persistent workgroups (1 per layer, 1024 thr = 32 waves),
//                        software-pipelined across layers via acquire/release flags.
//                        Per step: [64,256]x[256,768] bf16 WMMA GEMM(s), gates in LDS.
//  v2: n-tiles processed in pairs with immediate LDS spill (no scratch spills).
//  v3: explicit double-buffering of A/B fragments inside the K loops so loads
//      for kc+1 are in flight while WMMAs for kc execute (partial loadcnt waits).
// ---------------------------------------------------------------------------

typedef __bf16 bf16_t;
typedef __attribute__((ext_vector_type(16))) __bf16 v16bf;
typedef __attribute__((ext_vector_type(8)))  __bf16 v8bf;
typedef __attribute__((ext_vector_type(8)))  float  v8f;

#define B_  64
#define T_  1024
#define F_  16
#define H_  256
#define G_  768   // 3*H

// -------- workspace layout (bytes) --------
#define OFF_XP0   ((size_t)0)
#define SZ_XP0    ((size_t)B_*T_*G_*2)          // 96 MB
#define OFF_Y0    (OFF_XP0 + SZ_XP0)
#define SZ_Y      ((size_t)B_*T_*H_*2)          // 32 MB
#define OFF_Y1    (OFF_Y0 + SZ_Y)
#define OFF_UT0   (OFF_Y1 + SZ_Y)
#define SZ_MAT    ((size_t)H_*G_*2)             // 384 KB
#define OFF_WT1   (OFF_UT0 + SZ_MAT)
#define OFF_UT1   (OFF_WT1 + SZ_MAT)
#define OFF_WT2   (OFF_UT1 + SZ_MAT)
#define OFF_UT2   (OFF_WT2 + SZ_MAT)
#define OFF_BSUM  (OFF_UT2 + SZ_MAT)            // float[3*768]
#define OFF_BX    (OFF_BSUM + (size_t)3*G_*4)   // float[3*256]
#define OFF_FLAGS (OFF_BX   + (size_t)3*H_*4)   // uint[4]

// LDS pitches (elements) chosen for conflict-free b128 fragment loads
#define REC_PITCH 772   // f32 gate accumulators, 64 rows
#define XH_PITCH  260   // f32 x@W h-gate part, 64 rows
#define H_PITCH   264   // bf16 hidden state, 64 rows (528B row stride -> 4-bank lane shift)

union FragU { v16bf v; v8bf h[2]; };

__device__ __forceinline__ v16bf load_frag(const bf16_t* __restrict__ row, int ko) {
  // A/B fragment per ISA 16-bit layout: lane<16 -> K=ko..ko+7 & ko+16..ko+23,
  // lanes>=16 handled by caller adding +8 to ko.
  FragU f;
  f.h[0] = *(const v8bf*)(row + ko);
  f.h[1] = *(const v8bf*)(row + ko + 16);
  return f.v;
}

__device__ __forceinline__ v16bf load_frag_lds(const bf16_t* row, int ko) {
  FragU f;
  f.h[0] = *(const v8bf*)(row + ko);
  f.h[1] = *(const v8bf*)(row + ko + 16);
  return f.v;
}

__device__ __forceinline__ v8f wmma_bf16(v16bf a, v16bf b, v8f c) {
  return __builtin_amdgcn_wmma_f32_16x16x32_bf16(false, a, false, b, (short)0, c,
                                                 false, false);
}

__device__ __forceinline__ v8f vzero8() {
  v8f z;
  #pragma unroll
  for (int i = 0; i < 8; ++i) z[i] = 0.0f;
  return z;
}

// ---------------------------------------------------------------------------
// Phase 1: weight convert + transpose: src f32 [256][768] -> dst bf16 [768][256]
// ---------------------------------------------------------------------------
__global__ __launch_bounds__(256) void k_convert(
    const float* __restrict__ U0, const float* __restrict__ W1,
    const float* __restrict__ U1, const float* __restrict__ W2,
    const float* __restrict__ U2,
    bf16_t* __restrict__ Ut0, bf16_t* __restrict__ Wt1,
    bf16_t* __restrict__ Ut1, bf16_t* __restrict__ Wt2,
    bf16_t* __restrict__ Ut2) {
  int idx = blockIdx.x * 256 + threadIdx.x;
  if (idx >= 5 * H_ * G_) return;
  int m = idx / (H_ * G_);
  int e = idx % (H_ * G_);
  int r = e / G_;            // K (input dim)
  int c = e % G_;            // N (output dim)
  const float* src = (m == 0) ? U0 : (m == 1) ? W1 : (m == 2) ? U1 : (m == 3) ? W2 : U2;
  bf16_t*      dst = (m == 0) ? Ut0 : (m == 1) ? Wt1 : (m == 2) ? Ut1 : (m == 3) ? Wt2 : Ut2;
  dst[c * H_ + r] = (bf16_t)src[e];
}

// ---------------------------------------------------------------------------
// Phase 2: bias folding + flag reset (runs every call -> deterministic replay)
//   layer0: bsum = b0[1] (x-side bias is baked into Xp0), bx = 0
//   layer1/2: bsum[0..511] = b[0]+b[1]; bsum[512..767] = b[1]; bx = b[0][512+..]
// ---------------------------------------------------------------------------
__global__ __launch_bounds__(768) void k_bias(
    const float* __restrict__ b0, const float* __restrict__ b1,
    const float* __restrict__ b2, float* __restrict__ bsum,
    float* __restrict__ bx, unsigned int* __restrict__ flags) {
  int g = threadIdx.x;
  bsum[0 * G_ + g] = b0[G_ + g];
  bsum[1 * G_ + g] = (g < 2 * H_) ? (b1[g] + b1[G_ + g]) : b1[G_ + g];
  bsum[2 * G_ + g] = (g < 2 * H_) ? (b2[g] + b2[G_ + g]) : b2[G_ + g];
  if (g < H_) {
    bx[0 * H_ + g] = 0.0f;
    bx[1 * H_ + g] = b1[2 * H_ + g];
    bx[2 * H_ + g] = b2[2 * H_ + g];
  }
  if (g < 4) flags[g] = 0u;
}

// ---------------------------------------------------------------------------
// Phase 3: Xp0[bt][g] = sum_f float(batch[bt][f]) * W0[f][g] + b0[0][g]  (bf16 out)
// ---------------------------------------------------------------------------
#define EP_ROWS 16
__global__ __launch_bounds__(256) void k_embed(
    const int* __restrict__ batch, const float* __restrict__ W0,
    const float* __restrict__ b0, bf16_t* __restrict__ Xp0) {
  __shared__ float w[F_ * G_];        // 48 KB, L2-hot source
  __shared__ float x[EP_ROWS][F_];
  int tid = threadIdx.x;
  for (int i = tid; i < F_ * G_; i += 256) w[i] = W0[i];
  int bt0 = blockIdx.x * EP_ROWS;
  for (int i = tid; i < EP_ROWS * F_; i += 256)
    x[i / F_][i % F_] = (float)batch[(size_t)(bt0 + i / F_) * F_ + (i % F_)];
  __syncthreads();
  for (int r = 0; r < EP_ROWS; ++r) {
    #pragma unroll
    for (int cc = 0; cc < 3; ++cc) {
      int c = tid + cc * 256;
      float s = b0[c];
      #pragma unroll
      for (int f = 0; f < F_; ++f) s += x[r][f] * w[f * G_ + c];
      Xp0[(size_t)(bt0 + r) * G_ + c] = (bf16_t)s;
    }
  }
}

// ---------------------------------------------------------------------------
// Phase 4: pipelined scan. blockIdx.x = layer (0..2). 1024 threads = 32 waves.
// Tile map: 4 m-tiles x 48 n-tiles; wave w owns m = w/8, n-tiles (w%8)*6 .. +5,
// processed as 3 pairs (pairs never straddle the h-gate boundary n=32).
// ---------------------------------------------------------------------------
__global__ __launch_bounds__(1024, 1) void k_scan(
    const int* __restrict__ batch, const bf16_t* __restrict__ Xp0,
    bf16_t* __restrict__ Y0, bf16_t* __restrict__ Y1, float* __restrict__ out,
    const bf16_t* __restrict__ Ut0, const bf16_t* __restrict__ Wt1,
    const bf16_t* __restrict__ Ut1, const bf16_t* __restrict__ Wt2,
    const bf16_t* __restrict__ Ut2, const float* __restrict__ bsum,
    const float* __restrict__ bx, unsigned int* __restrict__ flags) {
  __shared__ float  rec[B_ * REC_PITCH];   // 197,632 B : x@W + h@U (z,r) / h@U (h-gate)
  __shared__ float  xh [B_ * XH_PITCH];    //  66,560 B : x@W h-gate part (layers 1,2)
  __shared__ bf16_t hb [B_ * H_PITCH];     //  33,792 B : hidden state, bf16
  __shared__ int    msk[B_];

  const int layer = blockIdx.x;
  const bf16_t* Ut    = (layer == 0) ? Ut0 : (layer == 1) ? Ut1 : Ut2;
  const bf16_t* Wt    = (layer == 1) ? Wt1 : Wt2;          // unused for layer 0
  const bf16_t* Yprev = (layer == 1) ? Y0  : Y1;           // unused for layer 0
  bf16_t*       Yout  = (layer == 0) ? Y0  : Y1;           // layer 2 -> d_out
  const float*  bs    = bsum + layer * G_;
  const float*  bxl   = bx   + layer * H_;

  const int tid  = threadIdx.x;
  const int wv   = tid >> 5;
  const int lane = tid & 31;
  const int m    = wv >> 3;               // 0..3
  const int n0   = (wv & 7) * 6;          // 0,6,..,42
  const int lrow = lane & 15;
  const int ksel = (lane >> 4) << 3;      // 0 | 8 : per-lane K sub-offset (ISA layout)

  const bf16_t* hrow = &hb[(m * 16 + lrow) * H_PITCH];

  // h0 = 0
  for (int i = tid; i < B_ * H_PITCH; i += 1024) hb[i] = (bf16_t)0.0f;
  __syncthreads();

  for (int t = 0; t < T_; ++t) {
    // ---- wait for producer layer's step t ----
    if (layer > 0) {
      if (tid == 0) {
        while (__hip_atomic_load(&flags[layer - 1], __ATOMIC_ACQUIRE,
                                 __HIP_MEMORY_SCOPE_AGENT) <= (unsigned)t) {
          __builtin_amdgcn_s_sleep(2);
        }
      }
    }
    __syncthreads();   // also orders prev-step h writes vs this step's reads

    if (tid < B_) msk[tid] = batch[(size_t)((tid << 10) | t) * F_ + (F_ - 1)];

    const bf16_t* arow =
        Yprev + ((size_t)(m * 16 + lrow) * T_ + t) * H_;   // y_prev row b, time t

    // ---- GEMMs, one pair of n-tiles at a time; spill to LDS immediately ----
    #pragma unroll
    for (int p = 0; p < 3; ++p) {
      const int ntA = n0 + 2 * p;
      const int ntB = ntA + 1;
      const bf16_t* bUa = Ut + (size_t)(ntA * 16 + lrow) * H_;
      const bf16_t* bUb = Ut + (size_t)(ntB * 16 + lrow) * H_;

      v8f accA = vzero8();
      v8f accB = vzero8();

      if (layer > 0) {
        // input GEMM: acc += y_prev[t] @ W   (double-buffered fragments)
        const bf16_t* bWa = Wt + (size_t)(ntA * 16 + lrow) * H_;
        const bf16_t* bWb = Wt + (size_t)(ntB * 16 + lrow) * H_;
        v16bf a_c  = load_frag(arow, ksel);
        v16bf bA_c = load_frag(bWa, ksel);
        v16bf bB_c = load_frag(bWb, ksel);
        #pragma unroll
        for (int kc = 0; kc < 8; ++kc) {
          v16bf a_n, bA_n, bB_n;
          if (kc < 7) {
            const int ko = (kc + 1) * 32 + ksel;
            a_n  = load_frag(arow, ko);
            bA_n = load_frag(bWa, ko);
            bB_n = load_frag(bWb, ko);
          }
          accA = wmma_bf16(a_c, bA_c, accA);
          accB = wmma_bf16(a_c, bB_c, accB);
          if (kc < 7) { a_c = a_n; bA_c = bA_n; bB_c = bB_n; }
        }
        // h-gate tiles (n>=32): x-part must stay separate (hh = tanh(xh + r*rh))
        if (ntA >= 32) {
          const int cA = (ntA - 32) * 16 + lrow;
          const int cB = (ntB - 32) * 16 + lrow;
          #pragma unroll
          for (int r = 0; r < 8; ++r) {
            xh[(m * 16 + r + ksel) * XH_PITCH + cA] = accA[r];
            xh[(m * 16 + r + ksel) * XH_PITCH + cB] = accB[r];
          }
          accA = vzero8();
          accB = vzero8();
        }
      }

      // recurrent GEMM: acc += h @ U  (A from LDS, B double-buffered from global)
      {
        v16bf a_c  = load_frag_lds(hrow, ksel);
        v16bf bA_c = load_frag(bUa, ksel);
        v16bf bB_c = load_frag(bUb, ksel);
        #pragma unroll
        for (int kc = 0; kc < 8; ++kc) {
          v16bf a_n, bA_n, bB_n;
          if (kc < 7) {
            const int ko = (kc + 1) * 32 + ksel;
            a_n  = load_frag_lds(hrow, ko);
            bA_n = load_frag(bUa, ko);
            bB_n = load_frag(bUb, ko);
          }
          accA = wmma_bf16(a_c, bA_c, accA);
          accB = wmma_bf16(a_c, bB_c, accB);
          if (kc < 7) { a_c = a_n; bA_c = bA_n; bB_c = bB_n; }
        }
      }

      // spill pair (C layout: VGPR r -> M = r + ksel, lane%16 -> N)
      const int cA = ntA * 16 + lrow;
      const int cB = ntB * 16 + lrow;
      #pragma unroll
      for (int r = 0; r < 8; ++r) {
        rec[(m * 16 + r + ksel) * REC_PITCH + cA] = accA[r];
        rec[(m * 16 + r + ksel) * REC_PITCH + cB] = accB[r];
      }
    }
    __syncthreads();

    // ---- gates: 16384 elements, 16 per thread ----
    #pragma unroll 4
    for (int k = 0; k < 16; ++k) {
      const int e = tid + (k << 10);
      const int b = e >> 8;
      const int j = e & (H_ - 1);
      const size_t bt = (size_t)((b << 10) | t);

      const float rz = rec[b * REC_PITCH + j]          + bs[j];
      const float rr = rec[b * REC_PITCH + H_ + j]     + bs[H_ + j];
      const float rh = rec[b * REC_PITCH + 2 * H_ + j] + bs[2 * H_ + j];

      float xz, xr, xhh;
      if (layer == 0) {
        const bf16_t* xp = Xp0 + bt * G_;
        xz  = (float)xp[j];
        xr  = (float)xp[H_ + j];
        xhh = (float)xp[2 * H_ + j];
      } else {
        xz  = 0.0f;
        xr  = 0.0f;
        xhh = xh[b * XH_PITCH + j] + bxl[j];
      }

      const float hprev = (float)hb[b * H_PITCH + j];
      const float z  = 1.0f / (1.0f + __expf(-(xz + rz)));
      const float rg = 1.0f / (1.0f + __expf(-(xr + rr)));
      const float hh = tanhf(xhh + rg * rh);
      float hn = z * hprev + (1.0f - z) * hh;
      hn = (msk[b] != -1) ? hn : hprev;          // Keras masking: carry state

      hb[b * H_PITCH + j] = (bf16_t)hn;
      if (layer < 2) {
        Yout[bt * H_ + j] = (bf16_t)hn;
      } else {
        out[bt * H_ + j] = hn;
        if (t == T_ - 1) out[(size_t)B_ * T_ * H_ + b * H_ + j] = hn;  // final state
      }
    }
    __syncthreads();

    // ---- publish step t to the next layer ----
    if (layer < 2 && tid == 0) {
      __hip_atomic_store(&flags[layer], (unsigned)(t + 1), __ATOMIC_RELEASE,
                         __HIP_MEMORY_SCOPE_AGENT);
    }
  }
}

// ---------------------------------------------------------------------------
extern "C" void kernel_launch(void* const* d_in, const int* in_sizes, int n_in,
                              void* d_out, int out_size, void* d_ws, size_t ws_size,
                              hipStream_t stream) {
  const int*   batch = (const int*)  d_in[0];
  const float* W0    = (const float*)d_in[1];
  const float* U0    = (const float*)d_in[2];
  const float* b0    = (const float*)d_in[3];
  const float* W1    = (const float*)d_in[4];
  const float* U1    = (const float*)d_in[5];
  const float* b1    = (const float*)d_in[6];
  const float* W2    = (const float*)d_in[7];
  const float* U2    = (const float*)d_in[8];
  const float* b2    = (const float*)d_in[9];
  float* out = (float*)d_out;

  char* ws = (char*)d_ws;
  bf16_t* Xp0  = (bf16_t*)(ws + OFF_XP0);
  bf16_t* Y0   = (bf16_t*)(ws + OFF_Y0);
  bf16_t* Y1   = (bf16_t*)(ws + OFF_Y1);
  bf16_t* Ut0  = (bf16_t*)(ws + OFF_UT0);
  bf16_t* Wt1  = (bf16_t*)(ws + OFF_WT1);
  bf16_t* Ut1  = (bf16_t*)(ws + OFF_UT1);
  bf16_t* Wt2  = (bf16_t*)(ws + OFF_WT2);
  bf16_t* Ut2  = (bf16_t*)(ws + OFF_UT2);
  float*  bsum = (float*) (ws + OFF_BSUM);
  float*  bx   = (float*) (ws + OFF_BX);
  unsigned int* flags = (unsigned int*)(ws + OFF_FLAGS);

  k_convert<<<(5 * H_ * G_ + 255) / 256, 256, 0, stream>>>(
      U0, W1, U1, W2, U2, Ut0, Wt1, Ut1, Wt2, Ut2);
  k_bias<<<1, 768, 0, stream>>>(b0, b1, b2, bsum, bx, flags);
  k_embed<<<(B_ * T_) / EP_ROWS, 256, 0, stream>>>(batch, W0, b0, Xp0);
  k_scan<<<3, 1024, 0, stream>>>(batch, Xp0, Y0, Y1, out,
                                 Ut0, Wt1, Ut1, Wt2, Ut2, bsum, bx, flags);
}
